// UnifiedAttention_84885733638357
// MI455X (gfx1250) — matrix operand order repaired and checked by reading, not compiled, not run on hardware
//
#include <hip/hip_runtime.h>

// ---------------------------------------------------------------------------
// MI455X (gfx1250) attention block: bf16 WMMA (f32 accumulate) everywhere.
//   B=2, S=2048, HIDDEN=2048, HEADS=16, HEAD_DIM=128
// ---------------------------------------------------------------------------

#define B_   2
#define S_   2048
#define H_   16
#define DH_  128
#define HID_ 2048

typedef __attribute__((ext_vector_type(16))) __bf16 bf16x16;
typedef __attribute__((ext_vector_type(8)))  float  f32x8;

union Frag {
  bf16x16 v;
  uint4   u[2];
};

// A-matrix fragment: 16x32 bf16, row-major source with leading dim `ld`.
// Lane L<16: row L, K = {k0..k0+7, k0+16..k0+23}; lane L+16: K = {+8..+15, +24..+31}.
static __device__ inline bf16x16 load_a_frag(const __bf16* __restrict__ base, int ld,
                                             int row0, int k0, int lane) {
  const int half = lane >> 4;
  const int r    = lane & 15;
  const __bf16* p = base + (size_t)(row0 + r) * ld + k0 + half * 8;
  Frag f;
  f.u[0] = *reinterpret_cast<const uint4*>(p);
  f.u[1] = *reinterpret_cast<const uint4*>(p + 16);
  return f.v;
}

// B-matrix fragment: 32x16 bf16 where B[k][n] = src[n0+n][k0+k] (src row-major, ld).
// Lane n: K = {k0..k0+7, k0+16..k0+23}; lane n+16: K = {+8..+15, +24..+31}.
static __device__ inline bf16x16 load_b_frag(const __bf16* __restrict__ base, int ld,
                                             int n0, int k0, int lane) {
  const int half = lane >> 4;
  const int n    = lane & 15;
  const __bf16* p = base + (size_t)(n0 + n) * ld + k0 + half * 8;
  Frag f;
  f.u[0] = *reinterpret_cast<const uint4*>(p);
  f.u[1] = *reinterpret_cast<const uint4*>(p + 16);
  return f.v;
}

static __device__ inline f32x8 wmma_bf16(bf16x16 a, bf16x16 b, f32x8 c) {
  // (neg_a, A, neg_b, B, c_mod, C, reuse_a, reuse_b)
  return __builtin_amdgcn_wmma_f32_16x16x32_bf16(false, a, false, b, (short)0, c,
                                                 false, false);
}

// ---------------------------------------------------------------------------
// fp32 -> bf16 conversion (4 elements / thread, b128 in, b64 out)
// ---------------------------------------------------------------------------
struct bf16x4s { __bf16 x, y, z, w; };

__global__ void cvt_f32_to_bf16(const float* __restrict__ in,
                                __bf16* __restrict__ out, long n4) {
  long i = (long)blockIdx.x * blockDim.x + threadIdx.x;
  if (i >= n4) return;
  float4 v = reinterpret_cast<const float4*>(in)[i];
  bf16x4s o;
  o.x = (__bf16)v.x; o.y = (__bf16)v.y; o.z = (__bf16)v.z; o.w = (__bf16)v.w;
  reinterpret_cast<bf16x4s*>(out)[i] = o;
}

// ---------------------------------------------------------------------------
// C[M,N](OutT) = A[M,K](bf16) @ W[N,K]^T(bf16) + bias[N](f32)
// Per-wave tile: 32M x 64N, K stepped by 32. 8 WMMAs per K step.
// __launch_bounds__(128, 1): fat VGPR budget, no accumulator spills — latency
// is hidden by clause-grouped b128 loads, not by wave occupancy.
// ---------------------------------------------------------------------------
template <typename OutT>
__global__ __launch_bounds__(128, 1) void
gemm_bf16_wmma(const __bf16* __restrict__ A,
               const __bf16* __restrict__ W,
               const float* __restrict__ bias,
               OutT* __restrict__ C, int M, int N, int K) {
  const int lane = threadIdx.x & 31;
  const int gw   = blockIdx.x * (blockDim.x >> 5) + (threadIdx.x >> 5);
  const int ntn  = N >> 6;
  const int tm   = gw / ntn;
  const int tn   = gw - tm * ntn;
  const int row0 = tm * 32;
  const int col0 = tn * 64;

  f32x8 acc[2][4] = {};

  for (int k0 = 0; k0 < K; k0 += 32) {
    const bf16x16 a0 = load_a_frag(A, K, row0,      k0, lane);
    const bf16x16 a1 = load_a_frag(A, K, row0 + 16, k0, lane);
#pragma unroll
    for (int j = 0; j < 4; j++) {
      const bf16x16 b = load_b_frag(W, K, col0 + j * 16, k0, lane);
      acc[0][j] = wmma_bf16(a0, b, acc[0][j]);
      acc[1][j] = wmma_bf16(a1, b, acc[1][j]);
    }
  }

  const int cl   = lane & 15;
  const int moff = (lane >> 4) * 8;
#pragma unroll
  for (int j = 0; j < 4; j++) {
    const int col = col0 + j * 16 + cl;
    const float bv = bias[col];
#pragma unroll
    for (int i = 0; i < 2; i++) {
#pragma unroll
      for (int r = 0; r < 8; r++) {
        const int row = row0 + i * 16 + moff + r;
        C[(size_t)row * N + col] = (OutT)(acc[i][j][r] + bv);
      }
    }
  }
}

// ---------------------------------------------------------------------------
// RoPE + layout: qkv[B,S,3,H,Dh] (bf16) ->
//   Qr[B,H,S,Dh], Kr[B,H,S,Dh] (rotated), Vt[B,H,Dh,S] (transposed for B-frags)
// One thread per (b,s,h,d) with d in [0,64): handles the (d, d+64) pair.
// ---------------------------------------------------------------------------
__global__ void rope_scatter(const __bf16* __restrict__ qkv,
                             __bf16* __restrict__ Qr, __bf16* __restrict__ Kr,
                             __bf16* __restrict__ Vt) {
  const int idx = blockIdx.x * blockDim.x + threadIdx.x;
  const int d = idx & 63;
  const int h = (idx >> 6) & (H_ - 1);
  const int s = (idx >> 10) & (S_ - 1);
  const int b = idx >> 21;

  const __bf16* row = qkv + (size_t)(b * S_ + s) * (3 * HID_);
  const int hd = h * DH_;
  const float q1 = (float)row[hd + d],            q2 = (float)row[hd + d + 64];
  const float k1 = (float)row[HID_ + hd + d],     k2 = (float)row[HID_ + hd + d + 64];
  const float v1 = (float)row[2 * HID_ + hd + d], v2 = (float)row[2 * HID_ + hd + d + 64];

  // inv_freq = 10000^(-2d/128) = exp(-d * ln(10000)/64)
  const float theta = (float)s * __expf((float)d * -0.14391156f);
  float sn, c;
  __sincosf(theta, &sn, &c);

  const size_t qo = ((size_t)(b * H_ + h) * S_ + s) * DH_;
  Qr[qo + d]      = (__bf16)(q1 * c - q2 * sn);
  Qr[qo + d + 64] = (__bf16)(q2 * c + q1 * sn);
  Kr[qo + d]      = (__bf16)(k1 * c - k2 * sn);
  Kr[qo + d + 64] = (__bf16)(k2 * c + k1 * sn);

  const size_t vo = (size_t)(b * H_ + h) * DH_;
  Vt[(vo + d) * S_ + s]      = (__bf16)v1;
  Vt[(vo + d + 64) * S_ + s] = (__bf16)v2;
}

// ---------------------------------------------------------------------------
// Flash attention, causal. One wave handles a 16-query tile of one (b,h).
// Key tiles of 32; online softmax; QK^T and P·V both via v_wmma bf16.
// P goes C-layout -> A-layout through a wave-private LDS slice.
// ---------------------------------------------------------------------------
#define PSTR 40  // 16 rows x 40 bf16 (80B row stride: conflict-free-ish, 16B aligned)

__global__ __launch_bounds__(128, 1) void
flash_attn(const __bf16* __restrict__ Qr,
           const __bf16* __restrict__ Kr,
           const __bf16* __restrict__ Vt,
           __bf16* __restrict__ AO) {
  __shared__ __align__(16) __bf16 plds[4 * 16 * PSTR];

  const int lane = threadIdx.x & 31;
  const int wid  = threadIdx.x >> 5;
  const int gw   = blockIdx.x * 4 + wid;   // 4 waves / block
  const int qt = gw & (S_ / 16 - 1);       // 128 query tiles
  const int h  = (gw >> 7) & (H_ - 1);
  const int b  = gw >> 11;

  const __bf16* Qh = Qr + (size_t)(b * H_ + h) * S_ * DH_;
  const __bf16* Kh = Kr + (size_t)(b * H_ + h) * S_ * DH_;
  const __bf16* Vh = Vt + (size_t)(b * H_ + h) * DH_ * S_;

  bf16x16 qf[4];
#pragma unroll
  for (int c = 0; c < 4; c++) qf[c] = load_a_frag(Qh, DH_, qt * 16, c * 32, lane);

  f32x8 acc[8] = {};
  float mrow[8], lrow[8];
#pragma unroll
  for (int r = 0; r < 8; r++) { mrow[r] = -1e30f; lrow[r] = 0.0f; }

  const int moff = (lane >> 4) * 8;  // C-layout: VGPR r holds row moff+r
  const int col  = lane & 15;
  __bf16* pl = plds + wid * 16 * PSTR;

  const int ktmax = (qt * 16 + 15) >> 5;
  for (int kt = 0; kt <= ktmax; kt++) {
    // S = Q K^T over Dh=128 (4 K-chunks x 2 N-subtiles = 8 WMMAs)
    f32x8 s0 = {}, s1 = {};
#pragma unroll
    for (int c = 0; c < 4; c++) {
      const bf16x16 kb0 = load_b_frag(Kh, DH_, kt * 32,      c * 32, lane);
      const bf16x16 kb1 = load_b_frag(Kh, DH_, kt * 32 + 16, c * 32, lane);
      s0 = wmma_bf16(qf[c], kb0, s0);
      s1 = wmma_bf16(qf[c], kb1, s1);
    }

    const float sc = 0.08838834764831845f;  // 1/sqrt(128)
    float corr[8];
#pragma unroll
    for (int r = 0; r < 8; r++) {
      const int qg  = qt * 16 + moff + r;
      const int kg0 = kt * 32 + col;
      float v0 = (kg0      <= qg) ? s0[r] * sc : -1e30f;  // causal mask
      float v1 = (kg0 + 16 <= qg) ? s1[r] * sc : -1e30f;
      // row max across the 16-lane half (xor<16 stays within the row group)
      float m = fmaxf(v0, v1);
#pragma unroll
      for (int off = 1; off < 16; off <<= 1) m = fmaxf(m, __shfl_xor(m, off, 32));
      const float mnew = fmaxf(mrow[r], m);
      corr[r] = __expf(mrow[r] - mnew);
      mrow[r] = mnew;
      const float e0 = __expf(v0 - mnew);
      const float e1 = __expf(v1 - mnew);
      s0[r] = e0; s1[r] = e1;
      float ps = e0 + e1;
#pragma unroll
      for (int off = 1; off < 16; off <<= 1) ps += __shfl_xor(ps, off, 32);
      lrow[r] = lrow[r] * corr[r] + ps;
    }
#pragma unroll
    for (int dt = 0; dt < 8; dt++)
#pragma unroll
      for (int r = 0; r < 8; r++) acc[dt][r] *= corr[r];

    // P: C-layout -> LDS -> A-layout (wave-private slice; DS ops in-order per wave)
#pragma unroll
    for (int r = 0; r < 8; r++) {
      pl[(moff + r) * PSTR + col]      = (__bf16)s0[r];
      pl[(moff + r) * PSTR + 16 + col] = (__bf16)s1[r];
    }
    asm volatile("s_wait_dscnt 0" ::: "memory");
    bf16x16 pf;
    {
      const int half = lane >> 4;
      const int rr   = lane & 15;
      Frag f;
      f.u[0] = *reinterpret_cast<const uint4*>(pl + rr * PSTR + half * 8);
      f.u[1] = *reinterpret_cast<const uint4*>(pl + rr * PSTR + 16 + half * 8);
      pf = f.v;
    }

    // O += P V : V^T layout makes B-frags contiguous. 8 WMMAs.
#pragma unroll
    for (int dt = 0; dt < 8; dt++) {
      const bf16x16 vb = load_b_frag(Vh, S_, dt * 16, kt * 32, lane);
      acc[dt] = wmma_bf16(pf, vb, acc[dt]);
    }
  }

  // normalize and store to AO[B,S,H*Dh] (bf16), ready as GEMM A operand
#pragma unroll
  for (int dt = 0; dt < 8; dt++) {
#pragma unroll
    for (int r = 0; r < 8; r++) {
      const int row = qt * 16 + moff + r;
      const float o = acc[dt][r] / lrow[r];
      AO[(size_t)(b * S_ + row) * HID_ + h * DH_ + dt * 16 + col] = (__bf16)o;
    }
  }
}

// ---------------------------------------------------------------------------
// Launch sequence
// ---------------------------------------------------------------------------
extern "C" void kernel_launch(void* const* d_in, const int* in_sizes, int n_in,
                              void* d_out, int out_size, void* d_ws, size_t ws_size,
                              hipStream_t stream) {
  const float* hs   = (const float*)d_in[0];
  const float* wqkv = (const float*)d_in[1];
  const float* bqkv = (const float*)d_in[2];
  const float* wd   = (const float*)d_in[3];
  const float* bd   = (const float*)d_in[4];
  float* out = (float*)d_out;

  char* ws = (char*)d_ws;
  size_t off = 0;
  auto take = [&](size_t elems) -> __bf16* {
    __bf16* p = (__bf16*)(ws + off);
    off += ((elems * sizeof(__bf16)) + 255) & ~(size_t)255;
    return p;
  };
  const size_t M = (size_t)B_ * S_;                 // 4096 rows
  __bf16* Abf = take(M * HID_);                     // hidden, bf16
  __bf16* Wq  = take((size_t)3 * HID_ * HID_);      // w_qkv, bf16
  __bf16* Wd  = take((size_t)HID_ * HID_);          // w_dense, bf16
  __bf16* QKV = take(M * 3 * HID_);                 // qkv projection out
  __bf16* Qr  = take(M * HID_);                     // Q rotated [B,H,S,Dh]
  __bf16* Kr  = take(M * HID_);                     // K rotated [B,H,S,Dh]
  __bf16* Vt  = take(M * HID_);                     // V transposed [B,H,Dh,S]
  __bf16* AO  = take(M * HID_);                     // attention out [B,S,HID]

  // 1) fp32 -> bf16
  cvt_f32_to_bf16<<<(int)(M * HID_ / 4 / 256), 256, 0, stream>>>(hs, Abf, (long)(M * HID_ / 4));
  cvt_f32_to_bf16<<<(int)(3UL * HID_ * HID_ / 4 / 256), 256, 0, stream>>>(wqkv, Wq, (long)(3UL * HID_ * HID_ / 4));
  cvt_f32_to_bf16<<<(int)(1UL * HID_ * HID_ / 4 / 256), 256, 0, stream>>>(wd, Wd, (long)(1UL * HID_ * HID_ / 4));

  // 2) QKV GEMM: [4096,2048] x [6144,2048]^T
  {
    const int waves = (int)(M / 32) * (3 * HID_ / 64);  // 12288
    gemm_bf16_wmma<__bf16><<<waves / 4, 128, 0, stream>>>(Abf, Wq, bqkv, QKV,
                                                          (int)M, 3 * HID_, HID_);
  }
  // 3) RoPE + layout
  rope_scatter<<<(B_ * S_ * H_ * 64) / 256, 256, 0, stream>>>(QKV, Qr, Kr, Vt);

  // 4) causal flash attention (one wave per 16-query tile)
  flash_attn<<<(B_ * H_ * (S_ / 16)) / 4, 128, 0, stream>>>(Qr, Kr, Vt, AO);

  // 5) output GEMM: [4096,2048] x [2048,2048]^T -> fp32 d_out
  {
    const int waves = (int)(M / 32) * (HID_ / 64);      // 4096
    gemm_bf16_wmma<float><<<waves / 4, 128, 0, stream>>>(AO, Wd, bd, out,
                                                         (int)M, HID_, HID_);
  }
}
